// GAUSS_24498493456490
// MI455X (gfx1250) — matrix operand order repaired and checked by reading, not compiled, not run on hardware
//
#include <hip/hip_runtime.h>
#include <math.h>

// ---------------- problem constants (from reference) ----------------
#define NN    1024
#define KNB   50
#define SREAL 51
#define SP    64      // padded ego size
#define FF    256
#define HID   512
#define CC    40
#define LAMP  10.0f
#define GLR   0.1f    // gamma/lambda
#define MAXIT 20
#define NPI   16      // subspace-iteration steps per ADMM iter
#define TOLR  1e-4f

typedef __bf16 bf16_t;
typedef __attribute__((ext_vector_type(16))) __bf16 v16bf;
typedef __attribute__((ext_vector_type(8)))  float  v8f;

// ---------------- WMMA fragment helpers (contiguous k = 16*(lane/16) + e, same for A and B) ----------------
// A fragment: A[m][k], 16x32 bf16, from LDS with row pitch `pitch` (elements).
// lane<16: M=lane,   K = {0..15}
// lane>=16:M=lane-16,K = {16..31}
__device__ inline v16bf load_frag_A(const bf16_t* lds, int pitch, int row0, int k0) {
  int lane = threadIdx.x & 31;
  int m    = row0 + (lane & 15);
  int kb   = (lane >> 4) * 16;                // 0 or 16
  const bf16_t* p = lds + m * pitch + k0;
  v16bf a;
#pragma unroll
  for (int e = 0; e < 8; ++e) a[e]     = p[kb + e];
#pragma unroll
  for (int e = 0; e < 8; ++e) a[8 + e] = p[kb + 8 + e];
  return a;
}

// B fragment from K-major storage: B[k][n] = lds[(k0+k)*pitch + col0+n], 32x16 bf16.
// VGPR v: lanes 0-15 hold K=2v,2v+1; lanes 16-31 hold K=16+2v,17+2v  => k(e)=e+(lane>=16?16:0)
__device__ inline v16bf load_frag_B_kn(const bf16_t* lds, int pitch, int k0, int col0) {
  int lane = threadIdx.x & 31;
  int n    = col0 + (lane & 15);
  int kb   = (lane >> 4) * 16;
  v16bf b;
#pragma unroll
  for (int e = 0; e < 16; ++e) b[e] = lds[(k0 + kb + e) * pitch + n];
  return b;
}

// B fragment when the logical B matrix is stored N-major (rows = n):
// B[k][n] = lds[(col0+n)*pitch + k0+k]   (used for Xe * Xe^T)
__device__ inline v16bf load_frag_B_nk(const bf16_t* lds, int pitch, int col0, int k0) {
  int lane = threadIdx.x & 31;
  int n    = col0 + (lane & 15);
  const bf16_t* p = lds + n * pitch + k0 + ((lane >> 4) * 16);
  v16bf b;
#pragma unroll
  for (int e = 0; e < 16; ++e) b[e] = p[e];
  return b;
}

// ---------------- Kernel 1: per-node ego ADMM (one block = one node) ----------------
__global__ __launch_bounds__(256) void ego_admm_kernel(
    const float* __restrict__ X, const int* __restrict__ nbr,
    float* __restrict__ H, bf16_t* __restrict__ Hbf) {

  __shared__ __align__(16) bf16_t sXe[SP * FF];   // 32KB; reused as S1/S2 after XXT
  __shared__ float sB[SP * SP];
  __shared__ float sXXT[SP * SP];
  __shared__ float sCH[SP * SP];
  __shared__ int   sEgo[SP];
  __shared__ float sDeg[SP];
  __shared__ float sU[SP * 3];
  __shared__ float sY[SP * 3];
  __shared__ float sRed[256];
  __shared__ float sG[16];
  __shared__ int   sFlag;

  const int nid = blockIdx.x;
  const int tid = threadIdx.x;

  if (tid < SP)
    sEgo[tid] = (tid == 0) ? nid : ((tid <= KNB) ? nbr[nid * KNB + tid - 1] : nid);
  if (tid == 0) sFlag = 0;
  __syncthreads();

  // load ego features (bf16) + init B = I (real block)
  for (int idx = tid; idx < SP * FF; idx += 256) {
    int r = idx >> 8;
    float v = (r < SREAL) ? X[(size_t)sEgo[r] * FF + (idx & (FF - 1))] : 0.0f;
    sXe[idx] = (bf16_t)v;
  }
  for (int idx = tid; idx < SP * SP; idx += 256) {
    int i = idx >> 6, j = idx & 63;
    sB[idx] = (i == j && i < SREAL) ? 1.0f : 0.0f;
  }
  __syncthreads();

  // XXT = Xe * Xe^T via WMMA bf16 -> f32, 4x4 grid of 16x16 tiles, 2 tiles/wave
  {
    int w = tid >> 5;
#pragma unroll
    for (int half = 0; half < 2; ++half) {
      int t  = w + half * 8;
      int ti = t >> 2, tj = t & 3;
      v8f c = {};
#pragma unroll
      for (int kb = 0; kb < FF; kb += 32) {
        v16bf a = load_frag_A   (sXe, FF, ti * 16, kb);
        v16bf b = load_frag_B_nk(sXe, FF, tj * 16, kb);
        c = __builtin_amdgcn_wmma_f32_16x16x32_bf16(false, a, false, b, (short)0, c, false, false);
      }
      int lane  = tid & 31;
      int col   = tj * 16 + (lane & 15);
      int rbase = ti * 16 + ((lane >> 4) ? 8 : 0);
#pragma unroll
      for (int r = 0; r < 8; ++r) sXXT[(rbase + r) * SP + col] = c[r];
    }
  }
  __syncthreads();

  // Msolve = XXT + lam*I (real block), identity on padding; Cholesky once.
  for (int idx = tid; idx < SP * SP; idx += 256) {
    int i = idx >> 6, j = idx & 63;
    sCH[idx] = (i < SREAL && j < SREAL) ? (sXXT[idx] + ((i == j) ? LAMP : 0.0f))
                                        : ((i == j) ? 1.0f : 0.0f);
  }
  __syncthreads();
  for (int k = 0; k < SP; ++k) {
    if (tid == 0) sCH[k * SP + k] = sqrtf(fmaxf(sCH[k * SP + k], 1e-20f));
    __syncthreads();
    for (int i = k + 1 + tid; i < SP; i += 256) sCH[i * SP + k] /= sCH[k * SP + k];
    __syncthreads();
    int n = SP - 1 - k;
    for (int idx = tid; idx < n * n; idx += 256) {
      int i = k + 1 + idx / n, j = k + 1 + idx % n;
      sCH[i * SP + j] -= sCH[i * SP + k] * sCH[j * SP + k];
    }
    __syncthreads();
  }

  float* S1 = (float*)sXe;          // 64x64 scratch (Laplacian, then W)
  float* S2 = S1 + SP * SP;         // 64x64 scratch (RHS -> Z)

  // ---------------- ADMM loop ----------------
  for (int it = 0; it < MAXIT; ++it) {
    if (sFlag) break;

    // Bs = clip((B+B^T)/2, 0) -> S1
    for (int idx = tid; idx < SP * SP; idx += 256) {
      int i = idx >> 6, j = idx & 63;
      S1[idx] = fmaxf(0.0f, 0.5f * (sB[i * SP + j] + sB[j * SP + i]));
    }
    __syncthreads();
    if (tid < SP) {
      float s = 0.0f;
      for (int j = 0; j < SP; ++j) s += S1[tid * SP + j];
      sDeg[tid] = s;
    }
    __syncthreads();
    if (tid == 0) {
      float mx = 0.0f;
      for (int r = 0; r < SP; ++r) mx = fmaxf(mx, sDeg[r]);
      sG[15] = 2.0f * mx + 1.0f;                       // Gershgorin shift c
    }
    __syncthreads();
    const float csh = sG[15];
    // L in-place over S1; padded rows get diag=c so (cI-L) kills them.
    for (int idx = tid; idx < SP * SP; idx += 256) {
      int i = idx >> 6, j = idx & 63;
      float l = ((i == j) ? sDeg[i] : 0.0f) - S1[idx];
      if (i >= SREAL || j >= SREAL) l = (i == j) ? csh : 0.0f;
      S1[idx] = l;
    }
    // deterministic subspace seed
    if (tid < SP * 3) {
      int i = tid / 3, k = tid % 3;
      sU[tid] = (i < SREAL) ? __sinf(0.37f * (float)((i + 1) * (k + 1)) + 0.11f * (float)(k + 1)) : 0.0f;
    }
    __syncthreads();

    // subspace iteration on (cI - L): top-3 subspace == smallest-3 of L
    for (int t = 0; t < NPI; ++t) {
      if (tid < SP * 3) {
        int i = tid / 3, k = tid % 3;
        float acc = csh * sU[i * 3 + k];
        for (int j = 0; j < SP; ++j) acc -= S1[i * SP + j] * sU[j * 3 + k];
        sY[i * 3 + k] = acc;
      }
      __syncthreads();
      if (tid < 6) sG[tid] = 0.0f;
      __syncthreads();
      if (tid < SP) {
        float y0 = sY[tid * 3], y1 = sY[tid * 3 + 1], y2 = sY[tid * 3 + 2];
        atomicAdd(&sG[0], y0 * y0); atomicAdd(&sG[1], y1 * y0); atomicAdd(&sG[2], y1 * y1);
        atomicAdd(&sG[3], y2 * y0); atomicAdd(&sG[4], y2 * y1); atomicAdd(&sG[5], y2 * y2);
      }
      __syncthreads();
      if (tid == 0) {                                   // chol(G)=R R^T, 3x3
        float r00 = sqrtf(sG[0] + 1e-12f);
        float r10 = sG[1] / r00, r20 = sG[3] / r00;
        float r11 = sqrtf(fmaxf(sG[2] - r10 * r10, 1e-20f));
        float r21 = (sG[4] - r20 * r10) / r11;
        float r22 = sqrtf(fmaxf(sG[5] - r20 * r20 - r21 * r21, 1e-20f));
        sG[8] = r00; sG[9] = r10; sG[10] = r11; sG[11] = r20; sG[12] = r21; sG[13] = r22;
      }
      __syncthreads();
      if (tid < SP) {                                   // U = Y R^{-T}
        float y0 = sY[tid * 3], y1 = sY[tid * 3 + 1], y2 = sY[tid * 3 + 2];
        float u0 = y0 / sG[8];
        float u1 = (y1 - sG[9] * u0) / sG[10];
        float u2 = (y2 - sG[11] * u0 - sG[12] * u1) / sG[13];
        sU[tid * 3] = u0; sU[tid * 3 + 1] = u1; sU[tid * 3 + 2] = u2;
      }
      __syncthreads();
    }

    // W = U U^T  (overwrites S1; L no longer needed)
    for (int idx = tid; idx < SP * SP; idx += 256) {
      int i = idx >> 6, j = idx & 63;
      S1[idx] = sU[i * 3] * sU[j * 3] + sU[i * 3 + 1] * sU[j * 3 + 1] + sU[i * 3 + 2] * sU[j * 3 + 2];
    }
    // RHS = XXT + lam*B (real block) -> S2
    for (int idx = tid; idx < SP * SP; idx += 256) {
      int i = idx >> 6, j = idx & 63;
      S2[idx] = (i < SREAL && j < SREAL) ? (sXXT[idx] + LAMP * sB[idx]) : 0.0f;
    }
    __syncthreads();
    // solve M Z = RHS with cached Cholesky, one column per thread
    if (tid < SP) {
      int j = tid;
      for (int i = 0; i < SP; ++i) {
        float s = S2[i * SP + j];
        for (int t2 = 0; t2 < i; ++t2) s -= sCH[i * SP + t2] * S2[t2 * SP + j];
        S2[i * SP + j] = s / sCH[i * SP + i];
      }
      for (int i = SP - 1; i >= 0; --i) {
        float s = S2[i * SP + j];
        for (int t2 = i + 1; t2 < SP; ++t2) s -= sCH[t2 * SP + i] * S2[t2 * SP + j];
        S2[i * SP + j] = s / sCH[i * SP + i];
      }
    }
    __syncthreads();

    // B update + convergence norm
    float dsq = 0.0f;
    for (int idx = tid; idx < SP * SP; idx += 256) {
      int i = idx >> 6, j = idx & 63;
      float bn = 0.0f;
      if (i != j && i < SREAL && j < SREAL) {
        float aij = S2[i * SP + j] - GLR * (S1[i * SP + i] - S1[i * SP + j]);
        float aji = S2[j * SP + i] - GLR * (S1[j * SP + j] - S1[j * SP + i]);
        bn = fmaxf(0.0f, 0.5f * (aij + aji));
      }
      float d = bn - sB[idx];
      dsq += d * d;
      sB[idx] = bn;
    }
    sRed[tid] = dsq;
    __syncthreads();
    for (int s = 128; s > 0; s >>= 1) {
      if (tid < s) sRed[tid] += sRed[tid + s];
      __syncthreads();
    }
    if (tid == 0 && sqrtf(sRed[0]) < TOLR) sFlag = 1;
    __syncthreads();
  }

  // propagate: H = (row-normalized B row 0) @ Xe
  if (tid == 0) {
    float s = 0.0f;
    for (int j = 0; j < SP; ++j) s += sB[j];
    sG[14] = (s > 0.0f) ? (1.0f / s) : 1.0f;
  }
  __syncthreads();
  if (tid < SP) sDeg[tid] = sB[tid] * sG[14];
  __syncthreads();
  float h = 0.0f;
  for (int j = 0; j < SREAL; ++j) h += sDeg[j] * X[(size_t)sEgo[j] * FF + tid];
  H[(size_t)nid * FF + tid]   = h;
  Hbf[(size_t)nid * FF + tid] = (bf16_t)h;
}

// ---------------- Kernel 2: f32 -> bf16 convert ----------------
__global__ void cvt_bf16_kernel(const float* __restrict__ in, bf16_t* __restrict__ out, int n) {
  int i = blockIdx.x * 256 + threadIdx.x;
  if (i < n) out[i] = (bf16_t)in[i];
}

// ---------------- Kernel 3: bf16 WMMA GEMM + bias + optional ReLU ----------------
// C[M,N] = act(A[M,K] @ B[K,N] + bias).  Block = 256 thr = 8 waves, macro-tile 32x64.
__global__ __launch_bounds__(256) void gemm_bias_act_kernel(
    const bf16_t* __restrict__ A, const bf16_t* __restrict__ Bw,
    const float* __restrict__ bias, float* __restrict__ outF, bf16_t* __restrict__ outB,
    int M, int N, int K, int relu) {
  __shared__ __align__(16) bf16_t As[32 * 32];
  __shared__ __align__(16) bf16_t Bs[32 * 64];

  const int m0 = blockIdx.y * 32;
  const int n0 = blockIdx.x * 64;
  const int tid = threadIdx.x;
  const int w = tid >> 5, ti = w & 1, tj = w >> 1;

  v8f c = {};
  for (int kb = 0; kb < K; kb += 32) {
    __syncthreads();
    {
      int r = tid >> 3, cg = (tid & 7) * 4;                   // 32x32 A chunk
#pragma unroll
      for (int e = 0; e < 4; ++e)
        As[r * 32 + cg + e] = A[(size_t)(m0 + r) * K + kb + cg + e];
    }
    {
      int r = tid >> 3, cg = (tid & 7) * 8;                   // 32x64 B chunk
#pragma unroll
      for (int e = 0; e < 8; ++e) {
        int n = n0 + cg + e;
        Bs[r * 64 + cg + e] = (n < N) ? Bw[(size_t)(kb + r) * N + n] : (bf16_t)0.0f;
      }
    }
    __syncthreads();
    v16bf a = load_frag_A   (As, 32, ti * 16, 0);
    v16bf b = load_frag_B_kn(Bs, 64, 0, tj * 16);
    c = __builtin_amdgcn_wmma_f32_16x16x32_bf16(false, a, false, b, (short)0, c, false, false);
  }

  int lane  = tid & 31;
  int col   = n0 + tj * 16 + (lane & 15);
  int rbase = m0 + ti * 16 + ((lane >> 4) ? 8 : 0);
  if (col < N) {
    float bv = bias[col];
#pragma unroll
    for (int r = 0; r < 8; ++r) {
      float v = c[r] + bv;
      if (relu) v = fmaxf(v, 0.0f);
      if (outF) outF[(size_t)(rbase + r) * N + col] = v;
      if (outB) outB[(size_t)(rbase + r) * N + col] = (bf16_t)v;
    }
  }
}

// ---------------- host ----------------
extern "C" void kernel_launch(void* const* d_in, const int* in_sizes, int n_in,
                              void* d_out, int out_size, void* d_ws, size_t ws_size,
                              hipStream_t stream) {
  const float* X   = (const float*)d_in[0];
  const int*   ei  = (const int*)d_in[1];
  const float* W1  = (const float*)d_in[2];
  const float* b1  = (const float*)d_in[3];
  const float* W2  = (const float*)d_in[4];
  const float* b2  = (const float*)d_in[5];
  const float* W3  = (const float*)d_in[6];
  const float* b3  = (const float*)d_in[7];
  float* out = (float*)d_out;
  (void)in_sizes; (void)n_in; (void)out_size; (void)ws_size;

  const int* nbr = ei + NN * KNB;   // edge_index row 1

  char* ws = (char*)d_ws;
  size_t off = 0;
  auto carve = [&](size_t bytes) { char* p = ws + off; off = (off + bytes + 255) & ~(size_t)255; return p; };
  float*  H    = (float*) carve((size_t)NN * FF * 4);
  bf16_t* Hbf  = (bf16_t*)carve((size_t)NN * FF * 2);
  bf16_t* W1b  = (bf16_t*)carve((size_t)FF * HID * 2);
  bf16_t* W2b  = (bf16_t*)carve((size_t)HID * HID * 2);
  bf16_t* W3b  = (bf16_t*)carve((size_t)HID * CC * 2);
  bf16_t* A1b  = (bf16_t*)carve((size_t)NN * HID * 2);
  bf16_t* A2b  = (bf16_t*)carve((size_t)NN * HID * 2);

  int n1 = FF * HID, n2 = HID * HID, n3 = HID * CC;
  cvt_bf16_kernel<<<(n1 + 255) / 256, 256, 0, stream>>>(W1, W1b, n1);
  cvt_bf16_kernel<<<(n2 + 255) / 256, 256, 0, stream>>>(W2, W2b, n2);
  cvt_bf16_kernel<<<(n3 + 255) / 256, 256, 0, stream>>>(W3, W3b, n3);

  ego_admm_kernel<<<NN, 256, 0, stream>>>(X, nbr, H, Hbf);

  gemm_bias_act_kernel<<<dim3(HID / 64, NN / 32), 256, 0, stream>>>(
      Hbf, W1b, b1, nullptr, A1b, NN, HID, FF, 1);
  gemm_bias_act_kernel<<<dim3(HID / 64, NN / 32), 256, 0, stream>>>(
      A1b, W2b, b2, nullptr, A2b, NN, HID, HID, 1);
  gemm_bias_act_kernel<<<dim3((CC + 63) / 64, NN / 32), 256, 0, stream>>>(
      A2b, W3b, b3, out, nullptr, NN, CC, HID, 0);
}
